// Model_80221399155101
// MI455X (gfx1250) — matrix operand order repaired and checked by reading, compiled, not run on hardware
//
#include <hip/hip_runtime.h>
#include <hip/hip_bf16.h>
#include <math.h>

typedef _Float16 v16h __attribute__((ext_vector_type(16)));
typedef _Float16 v8h  __attribute__((ext_vector_type(8)));
typedef float    v8f  __attribute__((ext_vector_type(8)));

#define RCUT 3.6f
#define SMW  0.3f
#define A_PITCH 40                 // halves per A row (80B, 16B aligned)
#define B_PITCH 32                 // halves per Bt row (64B)
#define PER_WAVE_BYTES (48*A_PITCH*2 + 16*B_PITCH*2 + 48*16*4) // 3840+1024+3072 = 7936
#define CONST_FLOATS 168           // 12 (0.5/sig^2) + 12 norm + 144 sinv

// ---------------------------------------------------------------------------
// Setup: compute sigma/norm and the 12x12 S^{-1/2} (Jacobi eigensolver) once
// per launch, on device, into d_ws.  Deterministic, graph-capture safe.
// ---------------------------------------------------------------------------
__global__ void soap_setup_kernel(float* __restrict__ ws) {
  if (threadIdx.x != 0 || blockIdx.x != 0) return;
  // gh[k] = Gamma(0.5*k), built by recurrence from Gamma(0.5)=sqrt(pi)
  double gh[27];
  gh[1] = 1.7724538509055160273;
  gh[2] = 1.0;
  for (int k = 3; k <= 26; ++k) gh[k] = 0.5 * (double)(k - 2) * gh[k - 2];

  double sigma[12], nrm[12];
  for (int n = 0; n < 12; ++n) {
    double s = sqrt((double)n); if (s < 1.0) s = 1.0;
    sigma[n] = 3.6 * s / 12.0;
    nrm[n]   = 1.0 / sqrt(0.5 * gh[2 * n + 3] * pow(sigma[n], 2.0 * n + 3.0));
  }
  // Overlap matrix S
  double A[12][12], V[12][12];
  for (int i = 0; i < 12; ++i)
    for (int j = 0; j < 12; ++j) {
      int npow = i + j;
      double c = 0.5 / (sigma[i] * sigma[i]) + 0.5 / (sigma[j] * sigma[j]);
      A[i][j] = nrm[i] * nrm[j] * 0.5 * gh[npow + 3] *
                pow(c, -((double)npow + 3.0) * 0.5);
      V[i][j] = (i == j) ? 1.0 : 0.0;
    }
  // Cyclic Jacobi
  for (int sweep = 0; sweep < 60; ++sweep) {
    for (int p = 0; p < 11; ++p)
      for (int q = p + 1; q < 12; ++q) {
        double apq = A[p][q];
        if (fabs(apq) < 1e-300) continue;
        double theta = 0.5 * (A[q][q] - A[p][p]) / apq;
        double t = ((theta >= 0.0) ? 1.0 : -1.0) /
                   (fabs(theta) + sqrt(theta * theta + 1.0));
        double c = 1.0 / sqrt(t * t + 1.0), s = t * c;
        for (int k = 0; k < 12; ++k) {
          double akp = A[k][p], akq = A[k][q];
          A[k][p] = c * akp - s * akq;
          A[k][q] = s * akp + c * akq;
        }
        for (int k = 0; k < 12; ++k) {
          double apk = A[p][k], aqk = A[q][k];
          A[p][k] = c * apk - s * aqk;
          A[q][k] = s * apk + c * aqk;
        }
        for (int k = 0; k < 12; ++k) {
          double vkp = V[k][p], vkq = V[k][q];
          V[k][p] = c * vkp - s * vkq;
          V[k][q] = s * vkp + c * vkq;
        }
      }
  }
  // ws layout: [0..11] 0.5/sigma^2, [12..23] norm, [24..167] Sinv_half (symmetric)
  for (int n = 0; n < 12; ++n) {
    ws[n]      = (float)(0.5 / (sigma[n] * sigma[n]));
    ws[12 + n] = (float)nrm[n];
  }
  for (int i = 0; i < 12; ++i)
    for (int j = 0; j < 12; ++j) {
      double s = 0.0;
      for (int k = 0; k < 12; ++k)
        s += V[i][k] * V[j][k] / sqrt(fabs(A[k][k]));
      ws[24 + i * 12 + j] = (float)s;
    }
}

// ---------------------------------------------------------------------------
// Main: one wave32 per atom; per-atom GEMM C(48x16) += A(48x32)*Y(32x16)
// via v_wmma_f32_16x16x32_f16, K-looped over 32 neighbor chunks.
// ---------------------------------------------------------------------------
__global__ __launch_bounds__(256)
void soap_main_kernel(const float* __restrict__ pos,
                      const int*   __restrict__ spec,
                      const float* __restrict__ Wg,
                      const float* __restrict__ bg,
                      const float* __restrict__ cst,
                      float* __restrict__ out, int total_atoms) {
  extern __shared__ char smem[];
  float* Wl   = (float*)smem;                 // 3456 floats
  float* cbuf = Wl + 3456;                    // 168 floats
  float* s2i  = cbuf;                         // 12
  float* nrm  = cbuf + 12;                    // 12
  float* sinv = cbuf + 24;                    // 144
  char*  wbase = (char*)(cbuf + CONST_FLOATS);

  const int tid  = threadIdx.x;
  const int lane = tid & 31;
  const int wave = tid >> 5;

  for (int t = tid; t < 3456; t += 256) Wl[t] = Wg[t];
  for (int t = tid; t < CONST_FLOATS; t += 256) cbuf[t] = cst[t];
  __syncthreads();

  const int atom = blockIdx.x * 8 + wave;
  if (atom >= total_atoms) return;
  const int bb = atom >> 10;
  const int ni = atom & 1023;

  char* my = wbase + wave * PER_WAVE_BYTES;
  _Float16* At = (_Float16*)my;                         // 48 x A_PITCH halves
  _Float16* Bt = (_Float16*)(my + 48 * A_PITCH * 2);    // 16 x B_PITCH halves
  float*    ct = (float*)(my + 48 * A_PITCH * 2 + 16 * B_PITCH * 2); // 48x16 f32

  const float* pbase = pos + (size_t)bb * 1024 * 3;
  const float xi = pbase[ni * 3 + 0];
  const float yi = pbase[ni * 3 + 1];
  const float zi = pbase[ni * 3 + 2];

  // zero the 12 pad rows of A once (never rewritten)
  for (int r = 36; r < 48; ++r) At[r * A_PITCH + lane] = (_Float16)0.0f;

  v8f acc0 = {0.f,0.f,0.f,0.f,0.f,0.f,0.f,0.f};
  v8f acc1 = acc0, acc2 = acc0;

  const int m  = lane & 15;
  const int ks = lane >> 4;

  for (int chunk = 0; chunk < 32; ++chunk) {
    const int j = chunk * 32 + lane;     // this lane's neighbor
    const float xj = pbase[j * 3 + 0];
    const float yj = pbase[j * 3 + 1];
    const float zj = pbase[j * 3 + 2];
    const int sj = spec[bb * 1024 + j];

    float dx = xi - xj, dy = yi - yj, dz = zi - zj;
    float d2 = dx * dx + dy * dy + dz * dz;
    const bool self = (j == ni);
    if (self) d2 = 1.0f;
    const float r   = sqrtf(d2);
    float tc = (r - (RCUT - SMW)) / SMW;
    tc = fminf(fmaxf(tc, 0.0f), 1.0f);
    float w = 0.5f * (1.0f + __cosf(3.14159265358979f * tc));
    if (self) w = 0.0f;
    const float inv = 1.0f / r;
    const float ux = dx * inv, uy = dy * inv, uz = dz * inv;

    // radial Gaussians (w folded in; SINV applied post-accumulation)
    float gw[12];
    {
      float p = w;
      #pragma unroll
      for (int n = 0; n < 12; ++n) {
        gw[n] = nrm[n] * p * __expf(-d2 * s2i[n]);
        p *= r;
      }
    }
    // A column: rows (species*12+n); other species rows zeroed
    #pragma unroll
    for (int a = 0; a < 3; ++a) {
      const bool hit = (a == sj);
      #pragma unroll
      for (int n = 0; n < 12; ++n)
        At[(a * 12 + n) * A_PITCH + lane] = (_Float16)(hit ? gw[n] : 0.0f);
    }
    // real spherical harmonics l<=3
    {
      const float x2 = ux * ux, y2 = uy * uy, z2 = uz * uz;
      float Y[16];
      Y[0]  = 0.28209479177387814f;
      Y[1]  = 0.4886025119029199f * uy;
      Y[2]  = 0.4886025119029199f * uz;
      Y[3]  = 0.4886025119029199f * ux;
      Y[4]  = 1.0925484305920792f * ux * uy;
      Y[5]  = 1.0925484305920792f * uy * uz;
      Y[6]  = 0.31539156525252005f * (3.0f * z2 - 1.0f);
      Y[7]  = 1.0925484305920792f * ux * uz;
      Y[8]  = 0.5462742152960396f * (x2 - y2);
      Y[9]  = 0.5900435899266435f * uy * (3.0f * x2 - y2);
      Y[10] = 2.890611442640554f * ux * uy * uz;
      Y[11] = 0.4570457994644658f * uy * (5.0f * z2 - 1.0f);
      Y[12] = 0.3731763325901154f * uz * (5.0f * z2 - 3.0f);
      Y[13] = 0.4570457994644658f * ux * (5.0f * z2 - 1.0f);
      Y[14] = 1.445305721320277f * uz * (x2 - y2);
      Y[15] = 0.5900435899266435f * ux * (3.0f * x2 - y2);
      #pragma unroll
      for (int k = 0; k < 16; ++k)
        Bt[k * B_PITCH + lane] = (_Float16)Y[k];
    }

    // same-wave LDS store -> WMMA operand load ordering (CDNA5 split counters)
    asm volatile("s_wait_dscnt 0" ::: "memory");

    // B operand: lane n = m, K halves {ks*8..+7, 16+ks*8..+7}
    v16h bmat;
    {
      const _Float16* bp = Bt + m * B_PITCH + ks * 8;
      v8h blo = *(const v8h*)(bp);
      v8h bhi = *(const v8h*)(bp + 16);
      #pragma unroll
      for (int i = 0; i < 8; ++i) { bmat[i] = blo[i]; bmat[8 + i] = bhi[i]; }
    }
    // A operand per row-tile: K halves {ks*8..+7, 16+ks*8..+7}
    #pragma unroll
    for (int t = 0; t < 3; ++t) {
      const _Float16* ar = At + (t * 16 + m) * A_PITCH;
      v8h alo = *(const v8h*)(ar + ks * 8);
      v8h ahi = *(const v8h*)(ar + 16 + ks * 8);
      v16h amat;
      #pragma unroll
      for (int i = 0; i < 8; ++i) { amat[i] = alo[i]; amat[8 + i] = ahi[i]; }
      if (t == 0)
        acc0 = __builtin_amdgcn_wmma_f32_16x16x32_f16(false, amat, false, bmat,
                                                      (short)0, acc0, false, false);
      else if (t == 1)
        acc1 = __builtin_amdgcn_wmma_f32_16x16x32_f16(false, amat, false, bmat,
                                                      (short)0, acc1, false, false);
      else
        acc2 = __builtin_amdgcn_wmma_f32_16x16x32_f16(false, amat, false, bmat,
                                                      (short)0, acc2, false, false);
    }
  }

  // spill C tiles (ISA layout: VGPR v -> row v (lanes 0-15) / v+8 (lanes 16-31))
  {
    const int colk = lane & 15;
    #pragma unroll
    for (int v = 0; v < 8; ++v) {
      const int ro = (lane < 16) ? v : v + 8;
      ct[(ro)      * 16 + colk] = acc0[v];
      ct[(16 + ro) * 16 + colk] = acc1[v];
      ct[(32 + ro) * 16 + colk] = acc2[v];
    }
  }
  asm volatile("s_wait_dscnt 0" ::: "memory");

  // apply S^{-1/2} on the radial index; overlay result on the dead A tile
  float* cf = (float*)my; // 36x16 floats fit in the 3840B A region
  for (int idx = lane; idx < 576; idx += 32) {
    const int r36 = idx >> 4, k = idx & 15;
    const int a = r36 / 12, n = r36 - a * 12;
    float s = 0.0f;
    #pragma unroll
    for (int v = 0; v < 12; ++v)
      s += sinv[v * 12 + n] * ct[(a * 12 + v) * 16 + k];
    cf[r36 * 16 + k] = s;
  }
  asm volatile("s_wait_dscnt 0" ::: "memory");

  // power spectrum folded directly into the W dot product
  const int   A1[6]   = {0, 0, 0, 1, 1, 2};
  const int   A2[6]   = {0, 1, 2, 1, 2, 2};
  const float FACT[6] = {1.0f, 1.41421356237f, 1.41421356237f,
                         1.0f, 1.41421356237f, 1.0f};
  const float IL[4]   = {1.0f, 0.57735026919f, 0.44721359549f, 0.37796447301f};
  float acc = 0.0f;
  for (int cb = lane; cb < 72; cb += 32) {
    const int pr = cb / 12, n = cb - pr * 12;
    const float* c1  = cf + (A1[pr] * 12 + n) * 16;
    const float* c2b = cf + A2[pr] * 12 * 16;
    const float* Wp  = Wl + (pr * 12 + n) * 48;
    const float fp = FACT[pr];
    for (int q = 0; q < 12; ++q) {
      const float* c2 = c2b + q * 16;
      const float s0 = c1[0] * c2[0];
      const float s1 = c1[1] * c2[1] + c1[2] * c2[2] + c1[3] * c2[3];
      const float s2 = c1[4] * c2[4] + c1[5] * c2[5] + c1[6] * c2[6] +
                       c1[7] * c2[7] + c1[8] * c2[8];
      const float s3 = c1[9] * c2[9] + c1[10] * c2[10] + c1[11] * c2[11] +
                       c1[12] * c2[12] + c1[13] * c2[13] + c1[14] * c2[14] +
                       c1[15] * c2[15];
      acc += fp * (s0 * IL[0] * Wp[q * 4 + 0] + s1 * IL[1] * Wp[q * 4 + 1] +
                   s2 * IL[2] * Wp[q * 4 + 2] + s3 * IL[3] * Wp[q * 4 + 3]);
    }
  }
  #pragma unroll
  for (int off = 16; off > 0; off >>= 1) acc += __shfl_down(acc, off, 32);
  if (lane == 0) out[atom] = acc + bg[0];
}

// ---------------------------------------------------------------------------
extern "C" void kernel_launch(void* const* d_in, const int* in_sizes, int n_in,
                              void* d_out, int out_size, void* d_ws, size_t ws_size,
                              hipStream_t stream) {
  const float* pos  = (const float*)d_in[0];  // (4,1024,3) f32
  const int*   spc  = (const int*)d_in[1];    // (4,1024) i32
  const float* W    = (const float*)d_in[2];  // (3456,) f32
  const float* b    = (const float*)d_in[3];  // (1,) f32
  float*       out  = (float*)d_out;          // (4096,1) f32
  float*       cst  = (float*)d_ws;

  soap_setup_kernel<<<1, 1, 0, stream>>>(cst);

  const int total_atoms = in_sizes[1];        // 4096
  const int blocks = (total_atoms + 7) / 8;   // 8 waves (atoms) per block
  const size_t shmem = 3456 * 4 + CONST_FLOATS * 4 + 8 * (size_t)PER_WAVE_BYTES;
  soap_main_kernel<<<blocks, 256, shmem, stream>>>(pos, spc, W, b, cst, out,
                                                   total_atoms);
}
